// MultiHeadAttention_45062796870012
// MI455X (gfx1250) — matrix operand order repaired and checked by reading, not compiled, not run on hardware
//
#include <hip/hip_runtime.h>
#include <hip/hip_bf16.h>
#include <type_traits>

// ---------------------------------------------------------------------------
// MHA forward for MI455X (gfx1250, wave32, WMMA).
// D_MODEL=1024, H=16, d_k=64, B=4, S=2048.
// ---------------------------------------------------------------------------

#define D_MODEL 1024
#define N_HEADS 16
#define D_KH    64
#define BATCH   4
#define SEQ     2048

typedef __bf16 bf16;
typedef __attribute__((ext_vector_type(16))) __bf16 v16bf;
typedef __attribute__((ext_vector_type(8)))  __bf16 v8bf;
typedef __attribute__((ext_vector_type(4)))  __bf16 v4bf;
typedef __attribute__((ext_vector_type(8)))  float  v8f;
typedef __attribute__((ext_vector_type(4)))  float  v4f;

__device__ __forceinline__ bf16 f2bf(float f) {
  // round-to-nearest-even fp32 -> bf16 via bit manipulation
  unsigned int u = __float_as_uint(f);
  u += 0x7FFFu + ((u >> 16) & 1u);
  unsigned short h = (unsigned short)(u >> 16);
  bf16 r;
  __builtin_memcpy(&r, &h, 2);
  return r;
}

__device__ __forceinline__ v8f wmma_bf16(v16bf a, v16bf b, v8f c) {
  return __builtin_amdgcn_wmma_f32_16x16x32_bf16(
      /*neg_a=*/false, a, /*neg_b=*/false, b,
      /*c_mod=*/(short)0, c, /*reuse_a=*/false, /*reuse_b=*/false);
}

// Async global->LDS staging (CDNA5 GLOBAL_LOAD_ASYNC_TO_LDS_B128, ASYNCcnt).
__device__ __forceinline__ void async_load_b128(void* lds_ptr, const void* gptr) {
  unsigned lds = (unsigned)(uintptr_t)lds_ptr;   // low 32b of flat LDS addr = LDS offset
  asm volatile("global_load_async_to_lds_b128 %0, %1, off"
               :: "v"(lds), "v"(gptr) : "memory");
}
__device__ __forceinline__ void wait_async() {
  asm volatile("s_wait_asynccnt 0x0" ::: "memory");
}
__device__ __forceinline__ void wait_lds() {
  asm volatile("s_wait_dscnt 0x0" ::: "memory");
}

// ---------------------------------------------------------------------------
// GEMM + bias:  C[M,N] = A[M,K] * W[K,N] + bias[N]
// Block: 256 threads (8 waves, 4x2), tile 128x128, K-step 32.
// Wave tile: 32x64 = 2x4 WMMA tiles of 16x16.
// ---------------------------------------------------------------------------
template <typename AT, typename OT>
__global__ __launch_bounds__(256) void gemm_bias_kernel(
    const AT* __restrict__ A, const float* __restrict__ W,
    const float* __restrict__ bias, OT* __restrict__ C,
    int M, int N, int K) {
  __shared__ alignas(32) bf16 As[128][32];   // [m][k]
  __shared__ alignas(32) bf16 Bs[128][32];   // [n][k]  (transposed for B-fragment)

  const int t    = threadIdx.x;
  const int lane = t & 31;
  const int w    = t >> 5;
  const int wm   = w & 3;
  const int wn   = w >> 2;
  const int half = lane >> 4;
  const int l16  = lane & 15;

  const int numBM = M >> 7;
  const int bm = (blockIdx.x % numBM) << 7;
  const int bn = (blockIdx.x / numBM) << 7;

  v8f acc[2][4];
#pragma unroll
  for (int i = 0; i < 2; ++i)
#pragma unroll
    for (int j = 0; j < 4; ++j)
#pragma unroll
      for (int e = 0; e < 8; ++e) acc[i][j][e] = 0.0f;

  for (int k0 = 0; k0 < K; k0 += 32) {
#pragma unroll
    for (int it = 0; it < 4; ++it) {
      int f   = t + it * 256;
      int row = f >> 3;
      int c4  = (f & 7) << 2;
      size_t off = (size_t)(bm + row) * K + k0 + c4;
      if constexpr (std::is_same<AT, float>::value) {
        v4f v = *(const v4f*)&A[off];
        As[row][c4 + 0] = f2bf(v.x);
        As[row][c4 + 1] = f2bf(v.y);
        As[row][c4 + 2] = f2bf(v.z);
        As[row][c4 + 3] = f2bf(v.w);
      } else {
        *(v4bf*)&As[row][c4] = *(const v4bf*)&A[off];
      }
    }
#pragma unroll
    for (int it = 0; it < 4; ++it) {
      int f  = t + it * 256;
      int kr = f >> 5;
      int n4 = (f & 31) << 2;
      v4f v = *(const v4f*)&W[(size_t)(k0 + kr) * N + bn + n4];
      Bs[n4 + 0][kr] = f2bf(v.x);
      Bs[n4 + 1][kr] = f2bf(v.y);
      Bs[n4 + 2][kr] = f2bf(v.z);
      Bs[n4 + 3][kr] = f2bf(v.w);
    }
    __syncthreads();

    v16bf af[2], bfr[4];
#pragma unroll
    for (int mt = 0; mt < 2; ++mt) {
      int row = wm * 32 + mt * 16 + l16;
      af[mt] = *(const v16bf*)&As[row][half * 16];
    }
#pragma unroll
    for (int nt = 0; nt < 4; ++nt) {
      int col = wn * 64 + nt * 16 + l16;
      bfr[nt] = *(const v16bf*)&Bs[col][half * 16];
    }
#pragma unroll
    for (int mt = 0; mt < 2; ++mt)
#pragma unroll
      for (int nt = 0; nt < 4; ++nt)
        acc[mt][nt] = wmma_bf16(af[mt], bfr[nt], acc[mt][nt]);
    __syncthreads();
  }

#pragma unroll
  for (int mt = 0; mt < 2; ++mt) {
    int rbase = bm + wm * 32 + mt * 16 + half * 8;
#pragma unroll
    for (int nt = 0; nt < 4; ++nt) {
      int col = bn + wn * 64 + nt * 16 + l16;
      float bv = bias[col];
#pragma unroll
      for (int v = 0; v < 8; ++v) {
        float val = acc[mt][nt][v] + bv;
        size_t off = (size_t)(rbase + v) * N + col;
        if constexpr (std::is_same<OT, float>::value) C[off] = val;
        else                                          C[off] = f2bf(val);
      }
    }
  }
}

// ---------------------------------------------------------------------------
// Flash attention (causal). Block = 256 thr = 8 waves; block owns 128 query
// rows of one (batch, head); wave owns 16 rows. Key loop in steps of 32.
// Softmax statistics are maintained in the A-fragment domain (one row per
// lane), so per-step cross-lane traffic is just 2 xor-16 shuffles plus a
// 16-float LDS transpose for the C-layout rescale factors.
// ---------------------------------------------------------------------------
__global__ __launch_bounds__(256) void flash_attn_kernel(
    const bf16* __restrict__ Qp, const bf16* __restrict__ Kp,
    const bf16* __restrict__ Vp, bf16* __restrict__ Ctx) {
  __shared__ alignas(32) bf16  Qs[128][D_KH];  // 16 KB
  __shared__ alignas(32) bf16  Ks[32][D_KH];   // 4 KB  [key][dk]
  __shared__ alignas(32) bf16  Vts[D_KH][32];  // 4 KB  [dk][key]
  __shared__ alignas(16) float Ss[8][16][32];  // 16 KB raw-score bounce (per wave)
  __shared__ alignas(16) float Al[8][16];      // per-row alpha / inv-l transpose

  const int t    = threadIdx.x;
  const int lane = t & 31;
  const int w    = t >> 5;
  const int half = lane >> 4;
  const int l16  = lane & 15;

  const int qtiles = SEQ / 128;
  int blk = blockIdx.x;
  int qb  = blk % qtiles;
  int bh  = blk / qtiles;
  int h   = bh % N_HEADS;
  int b   = bh / N_HEADS;
  int q0  = qb * 128;

  const size_t base = (size_t)b * SEQ * D_MODEL + (size_t)h * D_KH;

  // ---- stage Q tile 128x64 via async b128 (1024 chunks / 256 threads) ----
#pragma unroll
  for (int it = 0; it < 4; ++it) {
    int f   = t + it * 256;
    int row = f >> 3;                 // 8 chunks per 64-wide row
    int d8  = (f & 7) << 3;
    async_load_b128(&Qs[row][d8], &Qp[base + (size_t)(q0 + row) * D_MODEL + d8]);
  }
  wait_async();
  __syncthreads();

  // Q A-fragments (rows w*16..+15), dk chunks [0,32) and [32,64)
  v16bf aq[2];
  {
    int row = w * 16 + l16;
#pragma unroll
    for (int c = 0; c < 2; ++c) {
      aq[c] = *(const v16bf*)&Qs[row][c * 32 + half * 16];
    }
  }

  v8f acc[4];
#pragma unroll
  for (int nt = 0; nt < 4; ++nt)
#pragma unroll
    for (int e = 0; e < 8; ++e) acc[nt][e] = 0.0f;

  // A-layout softmax state: this lane owns row (w*16 + l16); the xor-16
  // partner lane holds the other half of the row's keys.
  float m_run = -1e30f, l_run = 0.0f;
  const int   qrow  = q0 + w * 16 + l16;   // row owned in A-layout domain
  const float scale = 0.125f;              // 1/sqrt(64)
  const int   kt_end = q0 + 128;

  for (int kt = 0; kt < kt_end; kt += 32) {
    __syncthreads();
    // ---- stage K tile (async, natural layout) ----
    {
      int row = t >> 3;
      int d8  = (t & 7) << 3;
      async_load_b128(&Ks[row][d8], &Kp[base + (size_t)(kt + row) * D_MODEL + d8]);
    }
    // ---- stage V tile (manual, transposed into B-fragment layout) ----
#pragma unroll
    for (int it = 0; it < 2; ++it) {
      int f   = t + it * 256;
      int row = f >> 4;
      int d4  = (f & 15) << 2;
      v4bf vv = *(const v4bf*)&Vp[base + (size_t)(kt + row) * D_MODEL + d4];
      Vts[d4 + 0][row] = vv[0];
      Vts[d4 + 1][row] = vv[1];
      Vts[d4 + 2][row] = vv[2];
      Vts[d4 + 3][row] = vv[3];
    }
    wait_async();
    __syncthreads();

    // ---- scores: S = Q * K^T (16 queries x 32 keys per wave) ----
    v8f s[2];
#pragma unroll
    for (int i = 0; i < 2; ++i)
#pragma unroll
      for (int e = 0; e < 8; ++e) s[i][e] = 0.0f;
#pragma unroll
    for (int nt = 0; nt < 2; ++nt)
#pragma unroll
      for (int c = 0; c < 2; ++c) {
        v16bf bk = *(const v16bf*)&Ks[nt * 16 + l16][c * 32 + half * 16];
        s[nt] = wmma_bf16(aq[c], bk, s[nt]);
      }

    // ---- bounce raw scores C-layout -> A-layout through LDS ----
#pragma unroll
    for (int v = 0; v < 8; ++v) {
      Ss[w][half * 8 + v][l16]      = s[0][v];
      Ss[w][half * 8 + v][16 + l16] = s[1][v];
    }
    wait_lds();

    // A-layout view: this lane's 16 keys of row (w*16+l16)
    v4f r0 = *(const v4f*)&Ss[w][l16][half * 8];
    v4f r1 = *(const v4f*)&Ss[w][l16][half * 8 + 4];
    v4f r2 = *(const v4f*)&Ss[w][l16][16 + half * 8];
    v4f r3 = *(const v4f*)&Ss[w][l16][16 + half * 8 + 4];
    float sv[16] = {r0.x, r0.y, r0.z, r0.w, r1.x, r1.y, r1.z, r1.w,
                    r2.x, r2.y, r2.z, r2.w, r3.x, r3.y, r3.z, r3.w};
    // mask + scale (key index for element i per A-fragment layout)
    float mx = -1e30f;
#pragma unroll
    for (int i = 0; i < 16; ++i) {
      int key = kt + ((i < 8) ? (half * 8 + i) : (16 + half * 8 + (i - 8)));
      sv[i] = (key <= qrow) ? sv[i] * scale : -1e30f;
      mx = fmaxf(mx, sv[i]);
    }
    mx = fmaxf(mx, __shfl_xor(mx, 16, 32));   // combine with row partner
    float mn    = fmaxf(m_run, mx);
    float alpha = __expf(m_run - mn);
    m_run = mn;

    v16bf ap;   // P A-fragment built directly in registers
    float rs = 0.0f;
#pragma unroll
    for (int i = 0; i < 16; ++i) {
      float p = __expf(sv[i] - mn);
      rs += p;
      ap[i] = f2bf(p);
    }
    rs += __shfl_xor(rs, 16, 32);
    l_run = l_run * alpha + rs;

    // ---- transpose alpha to C-layout rows via tiny LDS row ----
    if (half == 0) Al[w][l16] = alpha;
    wait_lds();
    v4f a0 = *(const v4f*)&Al[w][half * 8];
    v4f a1 = *(const v4f*)&Al[w][half * 8 + 4];
    float av[8] = {a0.x, a0.y, a0.z, a0.w, a1.x, a1.y, a1.z, a1.w};
#pragma unroll
    for (int nt = 0; nt < 4; ++nt)
#pragma unroll
      for (int v = 0; v < 8; ++v) acc[nt][v] *= av[v];

    // ---- ctx += P * V ----
#pragma unroll
    for (int nt = 0; nt < 4; ++nt) {
      v16bf bv;
      v8bf vlo = *(const v8bf*)&Vts[nt * 16 + l16][half * 8];
      v8bf vhi = *(const v8bf*)&Vts[nt * 16 + l16][16 + half * 8];
#pragma unroll
      for (int e = 0; e < 8; ++e) { bv[e] = vlo[e]; bv[8 + e] = vhi[e]; }
      acc[nt] = wmma_bf16(ap, bv, acc[nt]);
    }
  }

  // ---- normalize (1/l transposed to C-layout) and store ctx ----
  {
    float inv = 1.0f / l_run;
    if (half == 0) Al[w][l16] = inv;
    wait_lds();
    v4f i0 = *(const v4f*)&Al[w][half * 8];
    v4f i1 = *(const v4f*)&Al[w][half * 8 + 4];
    float iv[8] = {i0.x, i0.y, i0.z, i0.w, i1.x, i1.y, i1.z, i1.w};
#pragma unroll
    for (int v = 0; v < 8; ++v) {
      int r = q0 + w * 16 + half * 8 + v;
      size_t rowoff = base + (size_t)r * D_MODEL;
#pragma unroll
      for (int nt = 0; nt < 4; ++nt)
        Ctx[rowoff + nt * 16 + l16] = f2bf(acc[nt][v] * iv[v]);
    }
  }
}

// ---------------------------------------------------------------------------
extern "C" void kernel_launch(void* const* d_in, const int* in_sizes, int n_in,
                              void* d_out, int out_size, void* d_ws, size_t ws_size,
                              hipStream_t stream) {
  const float* q  = (const float*)d_in[0];
  const float* k  = (const float*)d_in[1];
  const float* v  = (const float*)d_in[2];
  // d_in[3] = causal mask (int32) -- causality handled analytically in-kernel
  const float* wq = (const float*)d_in[4];
  const float* bq = (const float*)d_in[5];
  const float* wk = (const float*)d_in[6];
  const float* bk = (const float*)d_in[7];
  const float* wv = (const float*)d_in[8];
  const float* bv = (const float*)d_in[9];
  const float* wo = (const float*)d_in[10];
  const float* bo = (const float*)d_in[11];
  float* out = (float*)d_out;

  const int M = BATCH * SEQ;   // 8192
  const int N = D_MODEL;       // 1024
  const int K = D_MODEL;       // 1024
  const size_t elems = (size_t)M * D_MODEL;

  bf16* Qp  = (bf16*)d_ws;
  bf16* Kp  = Qp + elems;
  bf16* Vp  = Kp + elems;
  bf16* Ctx = Vp + elems;      // total ws use: 4 * 16 MiB = 64 MiB

  dim3 blk(256);
  int gGemm = (M / 128) * (N / 128);                 // 512 blocks
  gemm_bias_kernel<float, bf16><<<gGemm, blk, 0, stream>>>(q, wq, bq, Qp, M, N, K);
  gemm_bias_kernel<float, bf16><<<gGemm, blk, 0, stream>>>(k, wk, bk, Kp, M, N, K);
  gemm_bias_kernel<float, bf16><<<gGemm, blk, 0, stream>>>(v, wv, bv, Vp, M, N, K);

  int gAttn = BATCH * N_HEADS * (SEQ / 128);         // 1024 blocks
  flash_attn_kernel<<<gAttn, blk, 0, stream>>>(Qp, Kp, Vp, Ctx);

  gemm_bias_kernel<bf16, float><<<gGemm, blk, 0, stream>>>(Ctx, wo, bo, out, M, N, K);
}
